// FlashStructureLayer_13168369729887
// MI455X (gfx1250) — matrix operand order repaired and checked by reading, not compiled, not run on hardware
//
#include <hip/hip_runtime.h>
#include <hip/hip_bf16.h>
#include <math.h>

typedef __attribute__((ext_vector_type(16))) _Float16 v16h;
typedef __attribute__((ext_vector_type(8)))  _Float16 v8h;
typedef __attribute__((ext_vector_type(8)))  float    v8f;
typedef __attribute__((ext_vector_type(4)))  unsigned u32x4;
typedef __attribute__((ext_vector_type(8)))  int      i32x8;
typedef __attribute__((ext_vector_type(4)))  int      i32x4;

#ifndef __has_builtin
#define __has_builtin(x) 0
#endif
#if __has_builtin(__builtin_amdgcn_tensor_load_to_lds) && \
    __has_builtin(__builtin_amdgcn_s_wait_tensorcnt)
#define HAVE_TDM 1
#else
#define HAVE_TDM 0
#endif
#if defined(__has_include)
#if __has_include(<hip/amd_detail/amd_gfx1250_TDM.h>)
#define TDM_SIX_ARGS 1
#endif
#endif
#ifndef TDM_SIX_ARGS
#define TDM_SIX_ARGS 0
#endif

namespace {
constexpr int B_ = 4, L_ = 1024, M_ = 4096, CS_ = 384, CP_ = 128, H_ = 8, C_ = 64;
constexpr int PQK_ = 8, PV_ = 12, RANK_ = 2, KNB_ = 10, CATD_ = 1408, FPAD_ = 288;
}

// ---------------------------------------------------------------------------
// WMMA helpers (CDNA5 layouts, wave32).
// A 16x32 f16: lane&15 = M row; lane>=16 selects K {8..15,24..31} half.
// B 32x16 f16: lane&15 = N col; lane>=16 selects K {8..15,24..31} half.
// C/D 16x16 f32: vgpr v = row (v or v+8 by lane half), lane&15 = col.
// ---------------------------------------------------------------------------
__device__ inline v8f wmma_f16(v16h a, v16h b, v8f c) {
  return __builtin_amdgcn_wmma_f32_16x16x32_f16(false, a, false, b, (short)0, c,
                                                false, false);
}

__device__ inline v16h load_a16(const _Float16* A, int lda, int m0, int k0) {
  int lane = threadIdx.x & 31;
  const _Float16* p = A + (size_t)(m0 + (lane & 15)) * lda + k0 + ((lane >> 4) << 3);
  v8h lo = *(const v8h*)p;
  v8h hi = *(const v8h*)(p + 16);
  v16h r;
#pragma unroll
  for (int i = 0; i < 8; ++i) { r[i] = lo[i]; r[i + 8] = hi[i]; }
  return r;
}

__device__ inline v16h load_b16(const _Float16* Bm, int ldb, int k0, int n0) {
  int lane = threadIdx.x & 31;
  const _Float16* p = Bm + (size_t)(k0 + ((lane >> 4) << 3)) * ldb + n0 + (lane & 15);
  v16h r;
#pragma unroll
  for (int i = 0; i < 8; ++i) { r[i] = p[(size_t)i * ldb]; r[i + 8] = p[(size_t)(16 + i) * ldb]; }
  return r;
}

// ---------------------------------------------------------------------------
// Generic WMMA GEMM: C[M,N] = A[M,K](f16) x B[K,N](f16) (+bias)(relu)(*rowscale)
// block = 128 threads = 4 waves; each wave computes a 16x64 strip (4 WMMAs per
// k-step off one A fragment); grid (M/64, ceil(N/64)). Software prefetch two
// k-steps ahead (global_prefetch_b8).
// ---------------------------------------------------------------------------
__global__ __launch_bounds__(128) void gemm_wmma(
    const _Float16* __restrict__ A, int lda,
    const _Float16* __restrict__ Bm, int ldb,
    const float* __restrict__ bias, const float* __restrict__ rowscale,
    float* __restrict__ Cf, _Float16* __restrict__ Ch, int ldc,
    int M, int N, int K, int relu) {
  int wave = threadIdx.x >> 5;
  int lane = threadIdx.x & 31;
  int m0 = (blockIdx.x << 6) + (wave << 4);
  int n0 = blockIdx.y << 6;
  if (m0 >= M) return;
  // clamp OOB subtiles onto a valid duplicate (loads stay in-bounds, store is
  // masked) so EXEC stays all-ones for the WMMAs.
  int nt[4];
#pragma unroll
  for (int t = 0; t < 4; ++t) {
    int n = n0 + (t << 4);
    nt[t] = (n < N) ? n : (N - 16);
  }
  v8f acc0 = {}, acc1 = {}, acc2 = {}, acc3 = {};
  for (int k0 = 0; k0 < K; k0 += 32) {
    if (k0 + 64 < K) {
      __builtin_prefetch(A + (size_t)(m0 + (lane & 15)) * lda + k0 + 64, 0, 1);
      __builtin_prefetch(Bm + (size_t)(k0 + 64 + lane) * ldb + n0, 0, 1);
    }
    v16h a = load_a16(A, lda, m0, k0);
    acc0 = wmma_f16(a, load_b16(Bm, ldb, k0, nt[0]), acc0);
    acc1 = wmma_f16(a, load_b16(Bm, ldb, k0, nt[1]), acc1);
    acc2 = wmma_f16(a, load_b16(Bm, ldb, k0, nt[2]), acc2);
    acc3 = wmma_f16(a, load_b16(Bm, ldb, k0, nt[3]), acc3);
  }
  int mb = m0 + ((lane >> 4) << 3);
#pragma unroll
  for (int t = 0; t < 4; ++t) {
    int nb = n0 + (t << 4);
    if (nb < N) {
      v8f acc = (t == 0) ? acc0 : (t == 1) ? acc1 : (t == 2) ? acc2 : acc3;
      int n = nb + (lane & 15);
      float bv = bias ? bias[n] : 0.0f;
#pragma unroll
      for (int v = 0; v < 8; ++v) {
        int row = mb + v;
        float x = acc[v] + bv;
        if (relu) x = fmaxf(x, 0.0f);
        if (rowscale) x *= rowscale[row];
        if (Cf) Cf[(size_t)row * ldc + n] = x;
        if (Ch) Ch[(size_t)row * ldc + n] = (_Float16)x;
      }
    }
  }
}

// ---------------------------------------------------------------------------
// Elementwise / pack kernels
// ---------------------------------------------------------------------------
__global__ void cvt_f32_f16(const float* __restrict__ x, _Float16* __restrict__ y,
                            long long n) {
  long long i = (long long)blockIdx.x * blockDim.x + threadIdx.x;
  if (i < n) y[i] = (_Float16)x[i];
}

__global__ void pack_wmlp(const float* __restrict__ W, _Float16* __restrict__ out) {
  int i = blockIdx.x * blockDim.x + threadIdx.x;
  if (i >= FPAD_ * CP_) return;
  int r = i / CP_, c = i % CP_;
  out[i] = (_Float16)((r < 2 * CP_ + 2 * KNB_) ? W[r * CP_ + c] : 0.0f);
}

// feat = [node(128) | pos_emb(128) | knn-dist(20) | pad(12)] as f16
__global__ void feat_kernel(const float* __restrict__ node,
                            const float* __restrict__ trans,
                            _Float16* __restrict__ feat) {
  long long i = (long long)blockIdx.x * blockDim.x + threadIdx.x;
  if (i >= (long long)M_ * FPAD_) return;
  int f = (int)(i % FPAD_);
  int m = (int)(i / FPAD_);
  int b = m >> 10, l = m & 1023;
  float val = 0.0f;
  if (f < CP_) {
    val = node[(size_t)m * CP_ + f];
  } else if (f < 2 * CP_) {
    int c = f - CP_;
    int cc = (c < 64) ? c : (c - 64);
    float freq = __expf(-9.210340371976184f * (1.0f / 64.0f) * (float)cc);
    float ang = (float)l * freq;
    val = (c < 64) ? __sinf(ang) : __cosf(ang);
  } else if (f < 2 * CP_ + 2 * KNB_) {
    int t = f - 2 * CP_;
    int off = (t < KNB_) ? (t - KNB_) : (t - KNB_ + 1);
    int j = l + off;
    j = j < 0 ? 0 : (j > L_ - 1 ? L_ - 1 : j);
    const float* ta = trans + ((size_t)b * L_ + j) * 3;
    const float* tb = trans + ((size_t)b * L_ + l) * 3;
    float dx = ta[0] - tb[0], dy = ta[1] - tb[1], dz = ta[2] - tb[2];
    val = sqrtf(dx * dx + dy * dy + dz * dz);
  }
  feat[i] = (_Float16)val;
}

// bf[m, r, h] = sum_c z[m, r, c] * Wb[c, h]
__global__ void bf_kernel(const float* __restrict__ z, const float* __restrict__ Wb,
                          float* __restrict__ out) {
  int idx = blockIdx.x * blockDim.x + threadIdx.x;
  if (idx >= M_ * RANK_ * H_) return;
  int h = idx & 7, r = (idx >> 3) & 1, m = idx >> 4;
  const float* zp = z + (size_t)m * (RANK_ * CP_) + r * CP_;
  float acc = 0.0f;
  for (int c = 0; c < CP_; ++c) acc += zp[c] * Wb[c * H_ + h];
  out[(size_t)m * 16 + r * 8 + h] = acc;
}

// kT[b,h,c,j] = kh[(b*L+j), h*64+c]
__global__ void kT_pack(const _Float16* __restrict__ kh, _Float16* __restrict__ kT) {
  long long i = (long long)blockIdx.x * blockDim.x + threadIdx.x;
  if (i >= (long long)B_ * H_ * 64 * L_) return;
  int j = (int)(i & 1023);
  long long r = i >> 10;
  int c = (int)(r & 63); r >>= 6;
  int h = (int)(r & 7);
  int b = (int)(r >> 3);
  kT[i] = kh[(((size_t)b * L_ + j) * (H_ * C_)) + h * C_ + c];
}

// rigid transform to global frame, build augmented point channels + row/col bias
__global__ void transform_kernel(
    const float* __restrict__ rots, const float* __restrict__ trans,
    const float* __restrict__ mask, const float* __restrict__ gamma,
    const float* __restrict__ qp_f, const float* __restrict__ kp_f,
    const float* __restrict__ vp_f, const float* __restrict__ bf1,
    const float* __restrict__ bf2, _Float16* __restrict__ paug,
    _Float16* __restrict__ kaugT, _Float16* __restrict__ vpg,
    float* __restrict__ rowbias, float* __restrict__ colbias) {
  int idx = blockIdx.x * blockDim.x + threadIdx.x;
  if (idx >= M_ * H_) return;
  int h = idx & 7, m = idx >> 3;
  int b = m >> 10, l = m & 1023;
  float R[9];
#pragma unroll
  for (int i = 0; i < 9; ++i) R[i] = rots[(size_t)m * 9 + i];
  float tx = trans[(size_t)m * 3 + 0] * 0.1f;
  float ty = trans[(size_t)m * 3 + 1] * 0.1f;
  float tz = trans[(size_t)m * 3 + 2] * 0.1f;
  float hw = log1pf(__expf(gamma[h])) * 0.09622504486493764f;  // sqrt(1/108)
  float sw = sqrtf(hw);
  const float c4 = 0.7598356856515925f;  // (1/3)^(1/4)
  // Q points
  float q2 = 0.0f;
  _Float16* pa = paug + ((size_t)m * H_ + h) * 32;
#pragma unroll
  for (int p = 0; p < PQK_; ++p) {
    const float* qq = qp_f + (size_t)m * (H_ * PQK_ * 3) + h * (PQK_ * 3) + p * 3;
    float gx = R[0] * qq[0] + R[1] * qq[1] + R[2] * qq[2] + tx;
    float gy = R[3] * qq[0] + R[4] * qq[1] + R[5] * qq[2] + ty;
    float gz = R[6] * qq[0] + R[7] * qq[1] + R[8] * qq[2] + tz;
    q2 += gx * gx + gy * gy + gz * gz;
    pa[p * 3 + 0] = (_Float16)(sw * gx);
    pa[p * 3 + 1] = (_Float16)(sw * gy);
    pa[p * 3 + 2] = (_Float16)(sw * gz);
  }
  pa[24] = (_Float16)(c4 * bf1[(size_t)m * 16 + h]);
  pa[25] = (_Float16)(c4 * bf1[(size_t)m * 16 + 8 + h]);
#pragma unroll
  for (int c = 26; c < 32; ++c) pa[c] = (_Float16)0.0f;
  rowbias[((size_t)(b * H_ + h)) * L_ + l] = -0.5f * hw * q2;
  // K points (transposed layout [32, L])
  float k2 = 0.0f;
  _Float16* ka = kaugT + ((size_t)(b * H_ + h)) * 32 * L_;
#pragma unroll
  for (int p = 0; p < PQK_; ++p) {
    const float* kk = kp_f + (size_t)m * (H_ * PQK_ * 3) + h * (PQK_ * 3) + p * 3;
    float gx = R[0] * kk[0] + R[1] * kk[1] + R[2] * kk[2] + tx;
    float gy = R[3] * kk[0] + R[4] * kk[1] + R[5] * kk[2] + ty;
    float gz = R[6] * kk[0] + R[7] * kk[1] + R[8] * kk[2] + tz;
    k2 += gx * gx + gy * gy + gz * gz;
    ka[(p * 3 + 0) * L_ + l] = (_Float16)(sw * gx);
    ka[(p * 3 + 1) * L_ + l] = (_Float16)(sw * gy);
    ka[(p * 3 + 2) * L_ + l] = (_Float16)(sw * gz);
  }
  ka[24 * L_ + l] = (_Float16)(c4 * bf2[(size_t)m * 16 + h]);
  ka[25 * L_ + l] = (_Float16)(c4 * bf2[(size_t)m * 16 + 8 + h]);
#pragma unroll
  for (int c = 26; c < 32; ++c) ka[c * L_ + l] = (_Float16)0.0f;
  colbias[((size_t)(b * H_ + h)) * L_ + l] =
      -0.5f * hw * k2 + (mask[m] - 1.0f) * 1e9f;
  // V points (global frame), padded 36 -> 48
  _Float16* vo = vpg + (size_t)m * (H_ * 48) + h * 48;
#pragma unroll
  for (int p = 0; p < PV_; ++p) {
    const float* vv = vp_f + (size_t)m * (H_ * PV_ * 3) + h * (PV_ * 3) + p * 3;
    float gx = R[0] * vv[0] + R[1] * vv[1] + R[2] * vv[2] + tx;
    float gy = R[3] * vv[0] + R[4] * vv[1] + R[5] * vv[2] + ty;
    float gz = R[6] * vv[0] + R[7] * vv[1] + R[8] * vv[2] + tz;
    vo[p * 3 + 0] = (_Float16)gx;
    vo[p * 3 + 1] = (_Float16)gy;
    vo[p * 3 + 2] = (_Float16)gz;
  }
#pragma unroll
  for (int c = 36; c < 48; ++c) vo[c] = (_Float16)0.0f;
}

// ---------------------------------------------------------------------------
// Flash IPA attention: 8 waves/block, each wave = 16 query rows of one (b,h).
// The 64KB augmented-K slab for this (b,h) is staged into LDS once per block
// via the Tensor Data Mover (TENSOR_LOAD_TO_LDS + s_wait_tensorcnt); per
// 32-col step: 6 score WMMAs (point+pair bias fused), online softmax, P staged
// via LDS into an A-fragment, 11 accumulate WMMAs (o, og, opair).
// ---------------------------------------------------------------------------
__global__ __launch_bounds__(256) void attn_kernel(
    const _Float16* __restrict__ qh, const _Float16* __restrict__ kT,
    const _Float16* __restrict__ vh, const _Float16* __restrict__ paug,
    const _Float16* __restrict__ kaugT, const _Float16* __restrict__ vpg,
    const _Float16* __restrict__ zv2, const float* __restrict__ rowbias,
    const float* __restrict__ colbias, float* __restrict__ o_f,
    float* __restrict__ og_f, float* __restrict__ opair_f) {
  __shared__ __align__(32) _Float16 lds_kaug[32 * L_];   // 64 KB slab
  __shared__ __align__(32) _Float16 lds_p[8][16 * 32];   // 8 KB P tiles
  const int wave = threadIdx.x >> 5;
  const int lane = threadIdx.x & 31;
  const int b = blockIdx.z, h = blockIdx.y;
  const int i0 = blockIdx.x * 128 + wave * 16;
  const int m0 = b * L_ + i0;
  const int mb = (lane >> 4) << 3;
  const int nc = lane & 15;
  const float inv_s = 0.07216878364870323f;  // 1/sqrt(3*64)

  const _Float16* kab = kaugT + ((size_t)(b * H_ + h)) * 32 * L_;
#if HAVE_TDM
  if (wave == 0) {
    // Tensor DMA descriptor (D#): 2D tensor 1024x32 of 2-byte elements,
    // tile = whole slab, contiguous into LDS.
    unsigned lds_addr = (unsigned)(size_t)(void*)&lds_kaug[0];
    unsigned long long ga = (unsigned long long)(size_t)kab;
    u32x4 g0 = {1u,                                   // count=1, load descriptor
                lds_addr,                             // lds_addr
                (unsigned)ga,                         // global_addr[31:0]
                (unsigned)((ga >> 32) & 0x01FFFFFFu) | (2u << 30)};  // type=2
    i32x8 g1 = {(int)(1u << 16),        // data_size=1 (2 bytes)
                (int)(1024u << 16),     // tensor_dim0 = 1024 (bits 79:48 lo)
                (int)(32u << 16),       // tensor_dim1 = 32   (bits 111:80 lo)
                (int)(1024u << 16),     // tile_dim0 = 1024   (bits 127:112)
                32,                     // tile_dim1 = 32     (bits 143:128)
                1024,                   // tensor_dim0_stride (bits 207:160)
                0, 0};
    i32x4 g2 = {0, 0, 0, 0};
    i32x4 g3 = {0, 0, 0, 0};
#if TDM_SIX_ARGS
    i32x8 g4 = {0, 0, 0, 0, 0, 0, 0, 0};
    __builtin_amdgcn_tensor_load_to_lds(g0, g1, g2, g3, g4, 0);
#else
    __builtin_amdgcn_tensor_load_to_lds(g0, g1, g2, g3, 0);
#endif
    __builtin_amdgcn_s_wait_tensorcnt(0);
  }
#else
  for (int idx = threadIdx.x; idx < 32 * L_; idx += 256)
    lds_kaug[idx] = kab[idx];
#endif
  __syncthreads();

  v16h aq0 = load_a16(qh, H_ * C_, m0, h * C_);
  v16h aq1 = load_a16(qh, H_ * C_, m0, h * C_ + 32);
  v16h apg = load_a16(paug, H_ * 32, m0, h * 32);

  float rb[8];
  {
    const float* rbp = rowbias + ((size_t)(b * H_ + h)) * L_ + i0 + mb;
#pragma unroll
    for (int v = 0; v < 8; ++v) rb[v] = rbp[v];
  }
  const float* cbp = colbias + ((size_t)(b * H_ + h)) * L_;
  const _Float16* kTb = kT + ((size_t)(b * H_ + h)) * 64 * L_;
  const _Float16* vb = vh + (size_t)b * L_ * (H_ * C_);
  const _Float16* vpb = vpg + (size_t)b * L_ * (H_ * 48);
  const _Float16* zvb = zv2 + (size_t)b * L_ * 64;

  v8f o0 = {}, o1 = {}, o2 = {}, o3 = {};
  v8f g0a = {}, g1a = {}, g2a = {};
  v8f p0 = {}, p1 = {}, p2 = {}, p3 = {};
  float mrun[8], lrun[8];
#pragma unroll
  for (int v = 0; v < 8; ++v) { mrun[v] = -1e30f; lrun[v] = 0.0f; }
  _Float16* pt = &lds_p[wave][0];

  for (int j0 = 0; j0 < L_; j0 += 32) {
    float sc[2][8];
#pragma unroll
    for (int t = 0; t < 2; ++t) {
      int n0 = j0 + t * 16;
      v8f acc = {};
      acc = wmma_f16(aq0, load_b16(kTb, L_, 0, n0), acc);
      acc = wmma_f16(aq1, load_b16(kTb, L_, 32, n0), acc);
      v8f accp = {};
      accp = wmma_f16(apg, load_b16(lds_kaug, L_, 0, n0), accp);
      float cb = cbp[n0 + nc];
#pragma unroll
      for (int v = 0; v < 8; ++v)
        sc[t][v] = acc[v] * inv_s + accp[v] + rb[v] + cb;
    }
    float scale[8];
#pragma unroll
    for (int v = 0; v < 8; ++v) {
      float mx = fmaxf(sc[0][v], sc[1][v]);
#pragma unroll
      for (int d = 1; d < 16; d <<= 1) mx = fmaxf(mx, __shfl_xor(mx, d, 32));
      float mnew = fmaxf(mrun[v], mx);
      scale[v] = __expf(mrun[v] - mnew);
      mrun[v] = mnew;
      sc[0][v] = __expf(sc[0][v] - mnew);
      sc[1][v] = __expf(sc[1][v] - mnew);
      float rs = sc[0][v] + sc[1][v];
#pragma unroll
      for (int d = 1; d < 16; d <<= 1) rs += __shfl_xor(rs, d, 32);
      lrun[v] = lrun[v] * scale[v] + rs;
      pt[(mb + v) * 32 + nc] = (_Float16)sc[0][v];
      pt[(mb + v) * 32 + 16 + nc] = (_Float16)sc[1][v];
      o0[v] *= scale[v]; o1[v] *= scale[v]; o2[v] *= scale[v]; o3[v] *= scale[v];
      g0a[v] *= scale[v]; g1a[v] *= scale[v]; g2a[v] *= scale[v];
      p0[v] *= scale[v]; p1[v] *= scale[v]; p2[v] *= scale[v]; p3[v] *= scale[v];
    }
    v16h ap;
    {
      const _Float16* pr = pt + (lane & 15) * 32 + ((lane >> 4) << 3);
#pragma unroll
      for (int i = 0; i < 8; ++i) { ap[i] = pr[i]; ap[i + 8] = pr[16 + i]; }
    }
    o0 = wmma_f16(ap, load_b16(vb, H_ * C_, j0, h * C_ + 0), o0);
    o1 = wmma_f16(ap, load_b16(vb, H_ * C_, j0, h * C_ + 16), o1);
    o2 = wmma_f16(ap, load_b16(vb, H_ * C_, j0, h * C_ + 32), o2);
    o3 = wmma_f16(ap, load_b16(vb, H_ * C_, j0, h * C_ + 48), o3);
    g0a = wmma_f16(ap, load_b16(vpb, H_ * 48, j0, h * 48 + 0), g0a);
    g1a = wmma_f16(ap, load_b16(vpb, H_ * 48, j0, h * 48 + 16), g1a);
    g2a = wmma_f16(ap, load_b16(vpb, H_ * 48, j0, h * 48 + 32), g2a);
    p0 = wmma_f16(ap, load_b16(zvb, 64, j0, 0), p0);
    p1 = wmma_f16(ap, load_b16(zvb, 64, j0, 16), p1);
    p2 = wmma_f16(ap, load_b16(zvb, 64, j0, 32), p2);
    p3 = wmma_f16(ap, load_b16(zvb, 64, j0, 48), p3);
  }
#pragma unroll
  for (int v = 0; v < 8; ++v) {
    float inv = 1.0f / lrun[v];
    size_t row = (size_t)(m0 + mb + v);
    float* op_ = o_f + row * (H_ * C_) + h * C_;
    op_[0 + nc] = o0[v] * inv; op_[16 + nc] = o1[v] * inv;
    op_[32 + nc] = o2[v] * inv; op_[48 + nc] = o3[v] * inv;
    float* gp_ = og_f + row * (H_ * 48) + h * 48;
    gp_[0 + nc] = g0a[v] * inv; gp_[16 + nc] = g1a[v] * inv; gp_[32 + nc] = g2a[v] * inv;
    float* pp_ = opair_f + row * (H_ * C_) + h * C_;
    pp_[0 + nc] = p0[v] * inv; pp_[16 + nc] = p1[v] * inv;
    pp_[32 + nc] = p2[v] * inv; pp_[48 + nc] = p3[v] * inv;
  }
}

// rotate og back to local frame, norms, opair = zv1 * (a@zv2), assemble cat f16
__global__ void cat_kernel(const float* __restrict__ o_f,
                           const float* __restrict__ og_f,
                           const float* __restrict__ opair_f,
                           const float* __restrict__ zv1,
                           const float* __restrict__ rots,
                           const float* __restrict__ trans,
                           _Float16* __restrict__ cat) {
  int idx = blockIdx.x * blockDim.x + threadIdx.x;
  if (idx >= M_ * H_) return;
  int h = idx & 7, m = idx >> 3;
  const float* R = rots + (size_t)m * 9;
  float tx = trans[(size_t)m * 3 + 0] * 0.1f;
  float ty = trans[(size_t)m * 3 + 1] * 0.1f;
  float tz = trans[(size_t)m * 3 + 2] * 0.1f;
  _Float16* cb = cat + (size_t)m * CATD_;
  const float* oo = o_f + (size_t)m * (H_ * C_) + h * C_;
  for (int c = 0; c < C_; ++c) cb[h * C_ + c] = (_Float16)oo[c];
  const float* gg = og_f + (size_t)m * (H_ * 48) + h * 48;
  for (int p = 0; p < PV_; ++p) {
    float dx = gg[p * 3 + 0] - tx, dy = gg[p * 3 + 1] - ty, dz = gg[p * 3 + 2] - tz;
    float ox = R[0] * dx + R[3] * dy + R[6] * dz;  // R^T
    float oy = R[1] * dx + R[4] * dy + R[7] * dz;
    float oz = R[2] * dx + R[5] * dy + R[8] * dz;
    cb[512 + h * (PV_ * 3) + p * 3 + 0] = (_Float16)ox;
    cb[512 + h * (PV_ * 3) + p * 3 + 1] = (_Float16)oy;
    cb[512 + h * (PV_ * 3) + p * 3 + 2] = (_Float16)oz;
    cb[800 + h * PV_ + p] = (_Float16)sqrtf(ox * ox + oy * oy + oz * oz + 1e-8f);
  }
  const float* pp = opair_f + (size_t)m * (H_ * C_) + h * C_;
  const float* z1v = zv1 + (size_t)m * 64;
  for (int d = 0; d < 64; ++d)
    cb[896 + h * 64 + d] = (_Float16)(z1v[d] * pp[d]);
}

// layernorm(resid + x) with gain/bias; one wave per row (CS=384 = 12*32)
__global__ __launch_bounds__(32) void ln_kernel(
    const float* __restrict__ x, const float* __restrict__ resid,
    const float* __restrict__ g, const float* __restrict__ bb,
    float* __restrict__ outf, _Float16* __restrict__ outh) {
  int m = blockIdx.x;
  int lane = threadIdx.x;
  float vals[12];
  float s = 0.0f;
#pragma unroll
  for (int i = 0; i < 12; ++i) {
    float v = x[(size_t)m * CS_ + i * 32 + lane] + resid[(size_t)m * CS_ + i * 32 + lane];
    vals[i] = v; s += v;
  }
#pragma unroll
  for (int d = 1; d < 32; d <<= 1) s += __shfl_xor(s, d, 32);
  float mean = s * (1.0f / CS_);
  float var = 0.0f;
#pragma unroll
  for (int i = 0; i < 12; ++i) { float dd = vals[i] - mean; var += dd * dd; }
#pragma unroll
  for (int d = 1; d < 32; d <<= 1) var += __shfl_xor(var, d, 32);
  var *= (1.0f / CS_);
  float r = rsqrtf(var + 1e-5f);
#pragma unroll
  for (int i = 0; i < 12; ++i) {
    int c = i * 32 + lane;
    float y = (vals[i] - mean) * r * g[c] + bb[c];
    outf[(size_t)m * CS_ + c] = y;
    if (outh) outh[(size_t)m * CS_ + c] = (_Float16)y;
  }
}

__global__ void u_kernel(const float* __restrict__ s2, const float* __restrict__ Wbb,
                         const float* __restrict__ bbb, float* __restrict__ u) {
  int idx = blockIdx.x * blockDim.x + threadIdx.x;
  if (idx >= M_ * 6) return;
  int k = idx % 6, m = idx / 6;
  float acc = bbb[k];
  const float* sp = s2 + (size_t)m * CS_;
  for (int c = 0; c < CS_; ++c) acc += sp[c] * Wbb[c * 6 + k];
  u[idx] = acc;
}

__global__ void compose_kernel(const float* __restrict__ u,
                               const float* __restrict__ rots,
                               const float* __restrict__ trans,
                               float* __restrict__ rout, float* __restrict__ tout) {
  int m = blockIdx.x * blockDim.x + threadIdx.x;
  if (m >= M_) return;
  const float* up = u + (size_t)m * 6;
  float qx = up[0], qy = up[1], qz = up[2];
  float inv = rsqrtf(1.0f + qx * qx + qy * qy + qz * qz);
  float w = inv, x = qx * inv, y = qy * inv, z = qz * inv;
  float Ru[9] = {1 - 2 * (y * y + z * z), 2 * (x * y - w * z), 2 * (x * z + w * y),
                 2 * (x * y + w * z), 1 - 2 * (x * x + z * z), 2 * (y * z - w * x),
                 2 * (x * z - w * y), 2 * (y * z + w * x), 1 - 2 * (x * x + y * y)};
  const float* R = rots + (size_t)m * 9;
  float* Ro = rout + (size_t)m * 9;
#pragma unroll
  for (int i = 0; i < 3; ++i)
#pragma unroll
    for (int j = 0; j < 3; ++j) {
      float acc = 0.0f;
#pragma unroll
      for (int k = 0; k < 3; ++k) acc += R[i * 3 + k] * Ru[k * 3 + j];
      Ro[i * 3 + j] = acc;
    }
  float tx = up[3], ty = up[4], tz = up[5];
  tout[(size_t)m * 3 + 0] = R[0] * tx + R[1] * ty + R[2] * tz + trans[(size_t)m * 3 + 0];
  tout[(size_t)m * 3 + 1] = R[3] * tx + R[4] * ty + R[5] * tz + trans[(size_t)m * 3 + 1];
  tout[(size_t)m * 3 + 2] = R[6] * tx + R[7] * ty + R[8] * tz + trans[(size_t)m * 3 + 2];
}

// ---------------------------------------------------------------------------
extern "C" void kernel_launch(void* const* d_in, const int* in_sizes, int n_in,
                              void* d_out, int out_size, void* d_ws, size_t ws_size,
                              hipStream_t stream) {
  (void)in_sizes; (void)n_in; (void)out_size; (void)ws_size;
  const float* s_in   = (const float*)d_in[0];
  const float* trans  = (const float*)d_in[1];
  const float* rots   = (const float*)d_in[2];
  const float* mask   = (const float*)d_in[3];
  const float* We_s   = (const float*)d_in[4];
  const float* be_s   = (const float*)d_in[5];
  const float* We_mlp = (const float*)d_in[6];
  const float* be_mlp = (const float*)d_in[7];
  const float* Wf1    = (const float*)d_in[8];
  const float* bf1b   = (const float*)d_in[9];
  const float* Wf2    = (const float*)d_in[10];
  const float* bf2b   = (const float*)d_in[11];
  const float* Wq     = (const float*)d_in[12];
  const float* Wk     = (const float*)d_in[13];
  const float* Wv     = (const float*)d_in[14];
  const float* Wqp    = (const float*)d_in[15];
  const float* Wkp    = (const float*)d_in[16];
  const float* Wvp    = (const float*)d_in[17];
  const float* gamma  = (const float*)d_in[18];
  const float* Wb1    = (const float*)d_in[19];
  const float* Wb2    = (const float*)d_in[20];
  const float* Wz1    = (const float*)d_in[21];
  const float* Wz2    = (const float*)d_in[22];
  const float* Wout   = (const float*)d_in[23];
  const float* bout   = (const float*)d_in[24];
  const float* ln1_g  = (const float*)d_in[25];
  const float* ln1_b  = (const float*)d_in[26];
  const float* Wt1    = (const float*)d_in[27];
  const float* bt1    = (const float*)d_in[28];
  const float* Wt2    = (const float*)d_in[29];
  const float* bt2    = (const float*)d_in[30];
  const float* Wt3    = (const float*)d_in[31];
  const float* bt3    = (const float*)d_in[32];
  const float* ln2_g  = (const float*)d_in[33];
  const float* ln2_b  = (const float*)d_in[34];
  const float* Wbb    = (const float*)d_in[35];
  const float* bbb    = (const float*)d_in[36];

  char* cur = (char*)d_ws;
  auto alloc = [&](size_t bytes) -> void* {
    void* p = (void*)cur;
    cur += (bytes + 255) & ~(size_t)255;
    return p;
  };
  // f16 buffers
  _Float16* sh     = (_Float16*)alloc((size_t)M_ * CS_ * 2);
  _Float16* wsh    = (_Float16*)alloc((size_t)CS_ * CP_ * 2);
  _Float16* wmlp_h = (_Float16*)alloc((size_t)FPAD_ * CP_ * 2);
  _Float16* wf1_h  = (_Float16*)alloc((size_t)CP_ * 256 * 2);
  _Float16* wf2_h  = (_Float16*)alloc((size_t)CP_ * 256 * 2);
  _Float16* wq_h   = (_Float16*)alloc((size_t)CS_ * 512 * 2);
  _Float16* wk_h   = (_Float16*)alloc((size_t)CS_ * 512 * 2);
  _Float16* wv_h   = (_Float16*)alloc((size_t)CS_ * 512 * 2);
  _Float16* wqp_h  = (_Float16*)alloc((size_t)CS_ * 192 * 2);
  _Float16* wkp_h  = (_Float16*)alloc((size_t)CS_ * 192 * 2);
  _Float16* wvp_h  = (_Float16*)alloc((size_t)CS_ * 288 * 2);
  _Float16* wz1_h  = (_Float16*)alloc((size_t)CP_ * 32 * 2);
  _Float16* wz2_h  = (_Float16*)alloc((size_t)CP_ * 32 * 2);
  _Float16* wout_h = (_Float16*)alloc((size_t)CATD_ * CS_ * 2);
  _Float16* wt1_h  = (_Float16*)alloc((size_t)CS_ * CS_ * 2);
  _Float16* wt2_h  = (_Float16*)alloc((size_t)CS_ * CS_ * 2);
  _Float16* wt3_h  = (_Float16*)alloc((size_t)CS_ * CS_ * 2);
  _Float16* feat_h = (_Float16*)alloc((size_t)M_ * FPAD_ * 2);
  _Float16* hmid_h = (_Float16*)alloc((size_t)M_ * CP_ * 2);
  _Float16* z1_h   = (_Float16*)alloc((size_t)M_ * 256 * 2);
  _Float16* z2_h   = (_Float16*)alloc((size_t)M_ * 256 * 2);
  _Float16* qh     = (_Float16*)alloc((size_t)M_ * 512 * 2);
  _Float16* kh     = (_Float16*)alloc((size_t)M_ * 512 * 2);
  _Float16* vh     = (_Float16*)alloc((size_t)M_ * 512 * 2);
  _Float16* kT_h   = (_Float16*)alloc((size_t)B_ * H_ * 64 * L_ * 2);
  _Float16* kaugT_h= (_Float16*)alloc((size_t)B_ * H_ * 32 * L_ * 2);
  _Float16* paug_h = (_Float16*)alloc((size_t)M_ * H_ * 32 * 2);
  _Float16* vpg_h  = (_Float16*)alloc((size_t)M_ * H_ * 48 * 2);
  _Float16* zv2_h  = (_Float16*)alloc((size_t)M_ * 64 * 2);
  _Float16* s1h    = (_Float16*)alloc((size_t)M_ * CS_ * 2);
  _Float16* h2a_h  = (_Float16*)alloc((size_t)M_ * CS_ * 2);
  _Float16* h2b_h  = (_Float16*)alloc((size_t)M_ * CS_ * 2);
  _Float16* cat_h  = (_Float16*)alloc((size_t)M_ * CATD_ * 2);
  // f32 buffers
  float* node_f  = (float*)alloc((size_t)M_ * CP_ * 4);
  float* z1_f    = (float*)alloc((size_t)M_ * 256 * 4);
  float* z2_f    = (float*)alloc((size_t)M_ * 256 * 4);
  float* qp_f    = (float*)alloc((size_t)M_ * 192 * 4);
  float* kp_f    = (float*)alloc((size_t)M_ * 192 * 4);
  float* vp_f    = (float*)alloc((size_t)M_ * 288 * 4);
  float* bf1_f   = (float*)alloc((size_t)M_ * 16 * 4);
  float* bf2_f   = (float*)alloc((size_t)M_ * 16 * 4);
  float* zv1_f   = (float*)alloc((size_t)M_ * 64 * 4);
  float* rowbias = (float*)alloc((size_t)B_ * H_ * L_ * 4);
  float* colbias = (float*)alloc((size_t)B_ * H_ * L_ * 4);
  float* o_f     = (float*)alloc((size_t)M_ * 512 * 4);
  float* og_f    = (float*)alloc((size_t)M_ * 384 * 4);
  float* opair_f = (float*)alloc((size_t)M_ * 512 * 4);
  float* attnout = (float*)alloc((size_t)M_ * CS_ * 4);
  float* s1_f    = (float*)alloc((size_t)M_ * CS_ * 4);
  float* t3_f    = (float*)alloc((size_t)M_ * CS_ * 4);
  float* u_f     = (float*)alloc((size_t)M_ * 6 * 4);

  auto cvt = [&](const float* src, _Float16* dst, long long n) {
    cvt_f32_f16<<<(unsigned)((n + 255) / 256), 256, 0, stream>>>(src, dst, n);
  };
  auto gemm = [&](const _Float16* A, int lda, const _Float16* Bm, int ldb,
                  const float* bias, const float* rowscale, float* Cf,
                  _Float16* Ch, int ldc, int Mn, int Nn, int Kn, int relu) {
    dim3 g((Mn + 63) / 64, (Nn + 63) / 64);
    gemm_wmma<<<g, 128, 0, stream>>>(A, lda, Bm, ldb, bias, rowscale, Cf, Ch,
                                     ldc, Mn, Nn, Kn, relu);
  };

  // pack inputs/weights to f16
  cvt(s_in, sh, (long long)M_ * CS_);
  cvt(We_s, wsh, (long long)CS_ * CP_);
  pack_wmlp<<<(FPAD_ * CP_ + 255) / 256, 256, 0, stream>>>(We_mlp, wmlp_h);
  cvt(Wf1, wf1_h, (long long)CP_ * 256);
  cvt(Wf2, wf2_h, (long long)CP_ * 256);
  cvt(Wq, wq_h, (long long)CS_ * 512);
  cvt(Wk, wk_h, (long long)CS_ * 512);
  cvt(Wv, wv_h, (long long)CS_ * 512);
  cvt(Wqp, wqp_h, (long long)CS_ * 192);
  cvt(Wkp, wkp_h, (long long)CS_ * 192);
  cvt(Wvp, wvp_h, (long long)CS_ * 288);
  cvt(Wz1, wz1_h, (long long)CP_ * 32);
  cvt(Wz2, wz2_h, (long long)CP_ * 32);
  cvt(Wout, wout_h, (long long)CATD_ * CS_);
  cvt(Wt1, wt1_h, (long long)CS_ * CS_);
  cvt(Wt2, wt2_h, (long long)CS_ * CS_);
  cvt(Wt3, wt3_h, (long long)CS_ * CS_);

  // edge embedder
  gemm(sh, CS_, wsh, CP_, be_s, nullptr, node_f, nullptr, CP_, M_, CP_, CS_, 0);
  feat_kernel<<<(unsigned)(((long long)M_ * FPAD_ + 255) / 256), 256, 0, stream>>>(
      node_f, trans, feat_h);
  gemm(feat_h, FPAD_, wmlp_h, CP_, be_mlp, nullptr, nullptr, hmid_h, CP_, M_, CP_, FPAD_, 1);
  gemm(hmid_h, CP_, wf1_h, 256, bf1b, mask, z1_f, z1_h, 256, M_, 256, CP_, 0);
  gemm(hmid_h, CP_, wf2_h, 256, bf2b, mask, z2_f, z2_h, 256, M_, 256, CP_, 0);

  // projections
  gemm(sh, CS_, wq_h, 512, nullptr, nullptr, nullptr, qh, 512, M_, 512, CS_, 0);
  gemm(sh, CS_, wk_h, 512, nullptr, nullptr, nullptr, kh, 512, M_, 512, CS_, 0);
  gemm(sh, CS_, wv_h, 512, nullptr, nullptr, nullptr, vh, 512, M_, 512, CS_, 0);
  gemm(sh, CS_, wqp_h, 192, nullptr, nullptr, qp_f, nullptr, 192, M_, 192, CS_, 0);
  gemm(sh, CS_, wkp_h, 192, nullptr, nullptr, kp_f, nullptr, 192, M_, 192, CS_, 0);
  gemm(sh, CS_, wvp_h, 288, nullptr, nullptr, vp_f, nullptr, 288, M_, 288, CS_, 0);

  // rank-2 pair bias factors and pair-value factors
  bf_kernel<<<(M_ * 16 + 255) / 256, 256, 0, stream>>>(z1_f, Wb1, bf1_f);
  bf_kernel<<<(M_ * 16 + 255) / 256, 256, 0, stream>>>(z2_f, Wb2, bf2_f);
  gemm(z1_h, 256, wz1_h, 32, nullptr, nullptr, zv1_f, nullptr, 64, M_, 32, CP_, 0);
  gemm(z1_h + CP_, 256, wz1_h, 32, nullptr, nullptr, zv1_f + 32, nullptr, 64, M_, 32, CP_, 0);
  gemm(z2_h, 256, wz2_h, 32, nullptr, nullptr, nullptr, zv2_h, 64, M_, 32, CP_, 0);
  gemm(z2_h + CP_, 256, wz2_h, 32, nullptr, nullptr, nullptr, zv2_h + 32, 64, M_, 32, CP_, 0);

  // attention operand prep
  kT_pack<<<(unsigned)(((long long)B_ * H_ * 64 * L_ + 255) / 256), 256, 0, stream>>>(kh, kT_h);
  transform_kernel<<<(M_ * H_ + 255) / 256, 256, 0, stream>>>(
      rots, trans, mask, gamma, qp_f, kp_f, vp_f, bf1_f, bf2_f, paug_h, kaugT_h,
      vpg_h, rowbias, colbias);

  // flash IPA attention (TDM-staged augmented-K slab in LDS)
  attn_kernel<<<dim3(L_ / 128, H_, B_), 256, 0, stream>>>(
      qh, kT_h, vh, paug_h, kaugT_h, vpg_h, zv2_h, rowbias, colbias, o_f, og_f,
      opair_f);

  // epilogue: cat + output projection + LN + transition + LN
  cat_kernel<<<(M_ * H_ + 255) / 256, 256, 0, stream>>>(o_f, og_f, opair_f, zv1_f,
                                                        rots, trans, cat_h);
  gemm(cat_h, CATD_, wout_h, CS_, bout, nullptr, attnout, nullptr, CS_, M_, CS_, CATD_, 0);
  ln_kernel<<<M_, 32, 0, stream>>>(attnout, s_in, ln1_g, ln1_b, s1_f, s1h);
  gemm(s1h, CS_, wt1_h, CS_, bt1, nullptr, nullptr, h2a_h, CS_, M_, CS_, CS_, 1);
  gemm(h2a_h, CS_, wt2_h, CS_, bt2, nullptr, nullptr, h2b_h, CS_, M_, CS_, CS_, 1);
  gemm(h2b_h, CS_, wt3_h, CS_, bt3, nullptr, t3_f, nullptr, CS_, M_, CS_, CS_, 0);
  float* s_out = (float*)d_out;
  ln_kernel<<<M_, 32, 0, stream>>>(t3_f, s1_f, ln2_g, ln2_b, s_out, nullptr);

  // backbone update + rigid compose
  u_kernel<<<(M_ * 6 + 255) / 256, 256, 0, stream>>>(s_out, Wbb, bbb, u_f);
  compose_kernel<<<(M_ + 255) / 256, 256, 0, stream>>>(
      u_f, rots, trans, s_out + (size_t)M_ * CS_,
      s_out + (size_t)M_ * CS_ + (size_t)M_ * 9);
}
